// UCBAttention_50096498540822
// MI455X (gfx1250) — compile-verified, matrix operand order checked by reading, not run on hardware
//
#include <hip/hip_runtime.h>
#include <stdint.h>

// ---------------- problem constants ----------------
#define Bz    16
#define Nn    577
#define Cc    768
#define Hh    12
#define HDm   64
#define BH    (Bz*Hh)        // 192
#define NPQ   592            // query tokens padded to mult of 16
#define NPK   608            // key tokens padded to mult of 32
#define NPK2  (NPK/2)        // 304 dwords per P row
#define KEEPK 144            // int(576*0.25)
#define ATT_SCALE 0.125f     // 64^-0.5
#define MROWS (Bz*Nn)        // 9232 (multiple of 16)

typedef __bf16 bf16;
typedef __attribute__((ext_vector_type(16))) __bf16 bf16x16;
typedef __attribute__((ext_vector_type(8)))  float  f32x8;
typedef __attribute__((ext_vector_type(4)))  uint32_t u32x4;
typedef __attribute__((ext_vector_type(4)))  int      i32x4;
typedef __attribute__((ext_vector_type(8)))  int      i32x8;

union FragU { uint32_t u[8]; bf16x16 v; };
union PairU { uint32_t u; bf16 h[2]; };

// dword (bf16-pair) index inside a 16-dword K-chunk for lane-group g, vgpr v
// (CDNA5 16-bit A/B layout: lanes0-15 K0-7/K16-23, lanes16-31 K8-15/K24-31)
__device__ __forceinline__ int kpOf(int g, int v) {
  return (v & 3) + 4*g + 8*(v >> 2);
}

__device__ __forceinline__ f32x8 wmma_bf16(const FragU& a, const FragU& b, f32x8 c) {
  return __builtin_amdgcn_wmma_f32_16x16x32_bf16(false, a.v, false, b.v, (short)0, c,
                                                 false, false);
}

// ---- Tensor Data Mover: 2D tile (4-byte elements) global -> LDS  ----
// D# per CDNA5 ISA. Groups 2/3 zero (<=2D tensor).
__device__ __forceinline__ void tdm_load_2d(uint32_t ldsOff, const void* gsrc,
                                            uint32_t tensorW, uint32_t tensorH,
                                            uint32_t tileW, uint32_t tileH,
                                            uint32_t strideElems)
{
  uint64_t ga = (uint64_t)(uintptr_t)gsrc;
  u32x4 g0;
  g0[0] = 1u;                                              // count=1 (valid user D#)
  g0[1] = ldsOff;                                          // lds_addr (bytes)
  g0[2] = (uint32_t)ga;                                    // global_addr[31:0]
  g0[3] = (uint32_t)((ga >> 32) & 0x1FFFFFFu) | (2u << 30);// addr[56:32] | type=2
  i32x8 g1;
  g1[0] = (int)(2u << 16);                                 // wg_mask=0, data_size=2 (4B)
  g1[1] = (int)((tensorW & 0xFFFFu) << 16);                // tensor_dim0[15:0]
  g1[2] = (int)((tensorW >> 16) | ((tensorH & 0xFFFFu) << 16)); // dim0 hi | dim1 lo
  g1[3] = (int)((tensorH >> 16) | (tileW << 16));          // dim1 hi | tile_dim0
  g1[4] = (int)(tileH & 0xFFFFu);                          // tile_dim1 (tile_dim2=0)
  g1[5] = (int)strideElems;                                // tensor_dim0_stride[31:0]
  g1[6] = 0;                                               // stride hi | dim1_stride lo
  g1[7] = 0;
  i32x4 g2; g2[0]=g2[1]=g2[2]=g2[3]=0;
  i32x4 g3; g3[0]=g3[1]=g3[2]=g3[3]=0;
#if defined(__clang_major__) && (__clang_major__ >= 23)
  i32x8 g4; g4[0]=g4[1]=g4[2]=g4[3]=g4[4]=g4[5]=g4[6]=g4[7]=0;
  __builtin_amdgcn_tensor_load_to_lds(g0, g1, g2, g3, g4, 0);
#else
  __builtin_amdgcn_tensor_load_to_lds(g0, g1, g2, g3, 0);
#endif
}

// ---------------- utility kernels ----------------
__global__ __launch_bounds__(256) void k_fill0(uint32_t* p, size_t n) {
  size_t i = (size_t)blockIdx.x*blockDim.x + threadIdx.x;
  size_t st = (size_t)gridDim.x*blockDim.x;
  for (; i < n; i += st) p[i] = 0u;
}

__global__ __launch_bounds__(256) void k_cvt_bf16(const float* __restrict__ s,
                                                  bf16* __restrict__ d, size_t n) {
  size_t i = (size_t)blockIdx.x*blockDim.x + threadIdx.x;
  size_t st = (size_t)gridDim.x*blockDim.x;
  for (; i < n; i += st) d[i] = (bf16)s[i];
}

// pack weights: wpk[kp][c] = (bf16(w[2kp][c]), bf16(w[2kp+1][c]))
__global__ __launch_bounds__(256) void k_cvt_pack(const float* __restrict__ w,
                                                  uint32_t* __restrict__ wpk,
                                                  int K2, int Nc) {
  size_t total = (size_t)K2 * Nc;
  size_t i = (size_t)blockIdx.x*blockDim.x + threadIdx.x;
  size_t st = (size_t)gridDim.x*blockDim.x;
  for (; i < total; i += st) {
    int kp = (int)(i / Nc), c = (int)(i % Nc);
    PairU pu;
    pu.h[0] = (bf16)w[(size_t)(2*kp  )*Nc + c];
    pu.h[1] = (bf16)w[(size_t)(2*kp+1)*Nc + c];
    wpk[i] = pu.u;
  }
}

// ---------------- QKV GEMM: X[9232,768] @ Wqkv[768,2304] -> q,k,vT (bf16) ----------------
// Block: 8 waves share one 64-column weight panel; per k-step the 16x64-dword
// B tile is DMA'd into LDS by the TDM, then read as ds_load fragments.
__global__ __launch_bounds__(256) void k_gemm_qkv(
    const bf16* __restrict__ xb, const uint32_t* __restrict__ wpk,
    bf16* __restrict__ qA, bf16* __restrict__ kA, bf16* __restrict__ vT)
{
  const int NG = (3*Cc)/64;                      // 36 column panels
  int ng  = blockIdx.x % NG;
  int mtg = blockIdx.x / NG;                     // 73 row-tile groups
  int wid = threadIdx.x >> 5;
  int lane = threadIdx.x & 31;
  int g = lane >> 4, ln = lane & 15;
  int mt = mtg*8 + wid;
  bool valid = (mt < 577);
  if (!valid) mt = 576;                          // clamp: compute, skip store

  __shared__ uint32_t ldsB[16*64];               // one k-step B tile (4KB)
  uint32_t ldsOff = (uint32_t)(uintptr_t)(&ldsB[0]);

  f32x8 acc0 = {}, acc1 = {}, acc2 = {}, acc3 = {};
  const uint32_t* arow = (const uint32_t*)xb + (size_t)(mt*16 + ln)*(Cc/2);

  for (int ks = 0; ks < Cc/32; ++ks) {
    __syncthreads();                             // previous tile fully consumed
    if (wid == 0) {
      const uint32_t* src = wpk + (size_t)(ks*16)*(3*Cc) + ng*64;
      tdm_load_2d(ldsOff, src, 3*Cc, (uint32_t)(Cc/2 - ks*16), 64, 16, 3*Cc);
      __builtin_amdgcn_s_wait_tensorcnt(0);
    }
    __syncthreads();                             // publish LDS tile

    __builtin_prefetch(arow + (ks+1)*16, 0, 1);
    FragU a;
#pragma unroll
    for (int v = 0; v < 8; ++v) a.u[v] = arow[ks*16 + kpOf(g, v)];
    FragU b0, b1, b2, b3;
#pragma unroll
    for (int v = 0; v < 8; ++v) {
      const uint32_t* lb = &ldsB[kpOf(g, v)*64 + ln];
      b0.u[v] = lb[0]; b1.u[v] = lb[16]; b2.u[v] = lb[32]; b3.u[v] = lb[48];
    }
    acc0 = wmma_bf16(a, b0, acc0);
    acc1 = wmma_bf16(a, b1, acc1);
    acc2 = wmma_bf16(a, b2, acc2);
    acc3 = wmma_bf16(a, b3, acc3);
  }

  if (valid) {
    f32x8 accs[4] = {acc0, acc1, acc2, acc3};
#pragma unroll
    for (int dt = 0; dt < 4; ++dt) {
      int col = ng*64 + dt*16 + ln;
      int which = col / Cc, c = col - which*Cc;
      int h = c >> 6, d = c & 63;
#pragma unroll
      for (int v = 0; v < 8; ++v) {
        int m = mt*16 + v + 8*g;
        int b = m / Nn, n = m - b*Nn;
        int bh = b*Hh + h;
        float val = accs[dt][v];
        if (which == 0)      qA[((size_t)bh*NPQ + n)*HDm + d] = (bf16)(val*ATT_SCALE);
        else if (which == 1) kA[((size_t)bh*NPK + n)*HDm + d] = (bf16)val;
        else                 vT[((size_t)bh*HDm + d)*NPK + n] = (bf16)val;
      }
    }
  }
}

// ---------------- pass1: S^T = K @ Q^T, softmax stats, store P=exp(S-m) bf16 ----------------
__global__ __launch_bounds__(256) void k_attn_pass1(
    const bf16* __restrict__ qA, const bf16* __restrict__ kA,
    bf16* __restrict__ P, float* __restrict__ lbuf)
{
  int task = blockIdx.x*8 + (threadIdx.x >> 5);
  if (task >= BH*37) return;
  int bh = task / 37, qt = task % 37;
  int lane = threadIdx.x & 31;
  int g = lane >> 4, ln = lane & 15;

  // B fragments = Q rows (this lane's column q = qt*16+ln)
  FragU bq0, bq1;
  const uint32_t* qrow = (const uint32_t*)qA + ((size_t)bh*NPQ + qt*16 + ln)*(HDm/2);
#pragma unroll
  for (int v = 0; v < 8; ++v) {
    int kp = kpOf(g, v);
    bq0.u[v] = qrow[kp]; bq1.u[v] = qrow[16 + kp];
  }

  const uint32_t* kb0 = (const uint32_t*)kA + (size_t)bh*NPK*(HDm/2);

  // sweep 1: running row max (per lane: its 8 k-positions of query q=ln)
  float rm = -1e30f;
  for (int jt = 0; jt < NPK/16; ++jt) {
    const uint32_t* krow = kb0 + (size_t)(jt*16 + ln)*(HDm/2);
    FragU a0, a1;
#pragma unroll
    for (int v = 0; v < 8; ++v) { int kp = kpOf(g, v); a0.u[v] = krow[kp]; a1.u[v] = krow[16+kp]; }
    f32x8 s = {};
    s = wmma_bf16(a0, bq0, s);
    s = wmma_bf16(a1, bq1, s);
    int kb = jt*16 + 8*g;
#pragma unroll
    for (int v = 0; v < 8; ++v) {
      float sv = (kb + v < Nn) ? s[v] : -1e30f;
      rm = fmaxf(rm, sv);
    }
  }
  rm = fmaxf(rm, __shfl_xor(rm, 16));

  // sweep 2: E = exp(S-m), accumulate l, store packed b128 rows of P
  float lsum = 0.f;
  for (int jt = 0; jt < NPK/16; ++jt) {
    const uint32_t* krow = kb0 + (size_t)(jt*16 + ln)*(HDm/2);
    FragU a0, a1;
#pragma unroll
    for (int v = 0; v < 8; ++v) { int kp = kpOf(g, v); a0.u[v] = krow[kp]; a1.u[v] = krow[16+kp]; }
    f32x8 s = {};
    s = wmma_bf16(a0, bq0, s);
    s = wmma_bf16(a1, bq1, s);
    int kb = jt*16 + 8*g;
    float e[8];
#pragma unroll
    for (int v = 0; v < 8; ++v) {
      float ev = (kb + v < Nn) ? __expf(s[v] - rm) : 0.f;
      lsum += ev;
      e[v] = ev;
    }
    PairU p0, p1, p2, p3;
    p0.h[0]=(bf16)e[0]; p0.h[1]=(bf16)e[1];
    p1.h[0]=(bf16)e[2]; p1.h[1]=(bf16)e[3];
    p2.h[0]=(bf16)e[4]; p2.h[1]=(bf16)e[5];
    p3.h[0]=(bf16)e[6]; p3.h[1]=(bf16)e[7];
    uint4 st; st.x=p0.u; st.y=p1.u; st.z=p2.u; st.w=p3.u;
    *(uint4*)(P + ((size_t)bh*NPQ + qt*16 + ln)*NPK + jt*16 + 8*g) = st;
  }
  lsum += __shfl_xor(lsum, 16);
  if (g == 0) lbuf[(size_t)bh*NPQ + qt*16 + ln] = lsum;
}

// ---------------- column sums of attn = P/l over (h,q) -> colsum[B,N] ----------------
__global__ __launch_bounds__(256) void k_colsum(
    const bf16* __restrict__ P, const float* __restrict__ lbuf, float* __restrict__ colsum)
{
  int blk = blockIdx.x;
  int bh = blk / 10, qc = blk % 10;
  int b = bh / Hh;
  int tid = threadIdx.x;
  float acc[3] = {0.f, 0.f, 0.f};
  int q0 = qc*64;
  int q1 = q0 + 64; if (q1 > Nn) q1 = Nn;
  for (int q = q0; q < q1; ++q) {
    float rl = 1.f / lbuf[(size_t)bh*NPQ + q];
    const bf16* row = P + ((size_t)bh*NPQ + q)*NPK;
#pragma unroll
    for (int i = 0; i < 3; ++i) {
      int c = tid + 256*i;
      if (c < Nn) acc[i] += (float)row[c] * rl;
    }
  }
#pragma unroll
  for (int i = 0; i < 3; ++i) {
    int c = tid + 256*i;
    if (c < Nn) atomicAdd(&colsum[(size_t)b*Nn + c], acc[i]);
  }
}

// ---------------- per-batch UCB top-k -> keep mask, packed bf16 bitmask, score_delta ----------------
__global__ __launch_bounds__(256) void k_topk(
    const float* __restrict__ colsum, const float* __restrict__ ucb_cnt,
    const int* __restrict__ counter, const int* __restrict__ enabled,
    float* __restrict__ keep, uint32_t* __restrict__ kmask, float* __restrict__ sdelta)
{
  int b = blockIdx.x, tid = threadIdx.x;
  __shared__ float su[NPQ];
  __shared__ float swv[8];
  __shared__ int   swi[8];
  bool prune = (enabled[0] != 0) && (counter[0] > 50);

  if (!prune) {
    for (int n = tid; n < NPQ; n += 256) keep[(size_t)b*NPQ + n] = (n < Nn) ? 1.f : 0.f;
    for (int n = tid; n < Nn;  n += 256) sdelta[(size_t)b*Nn + n] = 0.f;
    __syncthreads();
    for (int kp = tid; kp < NPK2; kp += 256) {
      int k0 = 2*kp, k1 = 2*kp + 1;
      uint32_t m = 0;
      if (k0 < Nn) m |= 0xFFFFu;
      if (k1 < Nn) m |= 0xFFFF0000u;
      kmask[(size_t)b*NPK2 + kp] = m;
    }
    return;
  }

  float lg = __logf((float)counter[0]);
  for (int n = tid; n < NPQ; n += 256) {
    float u;
    if (n == 0 || n >= Nn) u = -3e38f;
    else u = colsum[(size_t)b*Nn + n] * (1.f/(float)(Hh*Nn))
           + sqrtf(lg / (ucb_cnt[(size_t)b*Nn + n] + 1.f));
    su[n] = u;
    keep[(size_t)b*NPQ + n] = 0.f;
  }
  __syncthreads();

  for (int it = 0; it < KEEPK; ++it) {
    float bv = -3.3e38f; int bi = -1;
    for (int n = tid; n < Nn; n += 256) {
      float v = su[n];
      if (v > bv) { bv = v; bi = n; }
    }
    for (int off = 16; off > 0; off >>= 1) {
      float ov = __shfl_down(bv, off);
      int   oi = __shfl_down(bi, off);
      if (ov > bv) { bv = ov; bi = oi; }
    }
    if ((tid & 31) == 0) { swv[tid >> 5] = bv; swi[tid >> 5] = bi; }
    __syncthreads();
    if (tid == 0) {
      float mv = swv[0]; int mi = swi[0];
      for (int w = 1; w < 8; ++w) if (swv[w] > mv) { mv = swv[w]; mi = swi[w]; }
      su[mi] = -3e38f;
      keep[(size_t)b*NPQ + mi] = 1.f;
    }
    __syncthreads();
  }

  for (int n = tid; n < Nn; n += 256) sdelta[(size_t)b*Nn + n] = keep[(size_t)b*NPQ + n];
  __syncthreads();
  if (tid == 0) keep[(size_t)b*NPQ + 0] = 1.f;   // cls always kept
  __syncthreads();
  for (int kp = tid; kp < NPK2; kp += 256) {
    int k0 = 2*kp, k1 = 2*kp + 1;
    uint32_t m = 0;
    if (k0 < Nn && keep[(size_t)b*NPQ + k0] > 0.5f) m |= 0xFFFFu;
    if (k1 < Nn && keep[(size_t)b*NPQ + k1] > 0.5f) m |= 0xFFFF0000u;
    kmask[(size_t)b*NPK2 + kp] = m;
  }
}

// ---------------- masked PV: context = (P.mask) @ V / denom ----------------
__global__ __launch_bounds__(256) void k_av(
    const bf16* __restrict__ P, const float* __restrict__ lbuf,
    const uint32_t* __restrict__ kmask, const float* __restrict__ keep,
    const bf16* __restrict__ vT, bf16* __restrict__ ctx,
    const int* __restrict__ counter, const int* __restrict__ enabled)
{
  int task = blockIdx.x*8 + (threadIdx.x >> 5);
  if (task >= BH*37) return;
  int bh = task / 37, qt = task % 37;
  int b = bh / Hh, h = bh - b*Hh;
  int lane = threadIdx.x & 31, g = lane >> 4, ln = lane & 15;
  bool prune = (enabled[0] != 0) && (counter[0] > 50);
  int q = qt*16 + ln;
  bool qkept = !prune || (keep[(size_t)b*NPQ + q] > 0.5f);

  f32x8 acc0 = {}, acc1 = {}, acc2 = {}, acc3 = {};
  float rsum = 0.f;
  const uint32_t* prow  = (const uint32_t*)P + ((size_t)bh*NPQ + q)*NPK2;
  const uint32_t* kmrow = kmask + (size_t)b*NPK2;
  const uint32_t* vbase = (const uint32_t*)vT + (size_t)bh*HDm*NPK2;

  for (int j = 0; j < NPK/32; ++j) {
    FragU a;
#pragma unroll
    for (int v = 0; v < 8; ++v) {
      int kp = j*16 + kpOf(g, v);
      uint32_t pv = prow[kp];
      if (!qkept) pv &= kmrow[kp];       // zero pruned columns (bf16 +0)
      a.u[v] = pv;
      PairU pu; pu.u = pv;
      rsum += (float)pu.h[0] + (float)pu.h[1];
    }
    FragU b0, b1, b2, b3;
#pragma unroll
    for (int v = 0; v < 8; ++v) {
      int kp = j*16 + kpOf(g, v);
      b0.u[v] = vbase[(size_t)(ln     )*NPK2 + kp];
      b1.u[v] = vbase[(size_t)(16 + ln)*NPK2 + kp];
      b2.u[v] = vbase[(size_t)(32 + ln)*NPK2 + kp];
      b3.u[v] = vbase[(size_t)(48 + ln)*NPK2 + kp];
    }
    acc0 = wmma_bf16(a, b0, acc0);
    acc1 = wmma_bf16(a, b1, acc1);
    acc2 = wmma_bf16(a, b2, acc2);
    acc3 = wmma_bf16(a, b3, acc3);
  }
  rsum += __shfl_xor(rsum, 16);
  float lq = lbuf[(size_t)bh*NPQ + q];
  float denom = prune ? (rsum + lq*1e-8f) : lq;
  float rd = 1.f / denom;
  float rrec[8];
#pragma unroll
  for (int v = 0; v < 8; ++v) rrec[v] = __shfl(rd, v + 8*g);  // denom of row v+8g

  f32x8 accs[4] = {acc0, acc1, acc2, acc3};
#pragma unroll
  for (int dt = 0; dt < 4; ++dt) {
#pragma unroll
    for (int v = 0; v < 8; ++v) {
      int qq = qt*16 + v + 8*g;
      if (qq < Nn)
        ctx[((size_t)(b*Nn + qq))*Cc + h*HDm + dt*16 + ln] = (bf16)(accs[dt][v]*rrec[v]);
    }
  }
}

// ---------------- proj GEMM: ctx[9232,768] @ Wproj[768,768] + bias -> out f32 ----------------
__global__ __launch_bounds__(256) void k_gemm_proj(
    const bf16* __restrict__ ctx, const uint32_t* __restrict__ wpk,
    const float* __restrict__ bias, float* __restrict__ out)
{
  const int NG = Cc/64;                          // 12 column panels
  int ng  = blockIdx.x % NG;
  int mtg = blockIdx.x / NG;                     // 73 row-tile groups
  int wid = threadIdx.x >> 5;
  int lane = threadIdx.x & 31;
  int g = lane >> 4, ln = lane & 15;
  int mt = mtg*8 + wid;
  bool valid = (mt < 577);
  if (!valid) mt = 576;

  __shared__ uint32_t ldsB[16*64];
  uint32_t ldsOff = (uint32_t)(uintptr_t)(&ldsB[0]);

  f32x8 acc0 = {}, acc1 = {}, acc2 = {}, acc3 = {};
  const uint32_t* arow = (const uint32_t*)ctx + (size_t)(mt*16 + ln)*(Cc/2);

  for (int ks = 0; ks < Cc/32; ++ks) {
    __syncthreads();
    if (wid == 0) {
      const uint32_t* src = wpk + (size_t)(ks*16)*Cc + ng*64;
      tdm_load_2d(ldsOff, src, Cc, (uint32_t)(Cc/2 - ks*16), 64, 16, Cc);
      __builtin_amdgcn_s_wait_tensorcnt(0);
    }
    __syncthreads();

    __builtin_prefetch(arow + (ks+1)*16, 0, 1);
    FragU a;
#pragma unroll
    for (int v = 0; v < 8; ++v) a.u[v] = arow[ks*16 + kpOf(g, v)];
    FragU b0, b1, b2, b3;
#pragma unroll
    for (int v = 0; v < 8; ++v) {
      const uint32_t* lb = &ldsB[kpOf(g, v)*64 + ln];
      b0.u[v] = lb[0]; b1.u[v] = lb[16]; b2.u[v] = lb[32]; b3.u[v] = lb[48];
    }
    acc0 = wmma_bf16(a, b0, acc0);
    acc1 = wmma_bf16(a, b1, acc1);
    acc2 = wmma_bf16(a, b2, acc2);
    acc3 = wmma_bf16(a, b3, acc3);
  }

  if (valid) {
    f32x8 accs[4] = {acc0, acc1, acc2, acc3};
#pragma unroll
    for (int dt = 0; dt < 4; ++dt) {
      int col = ng*64 + dt*16 + ln;
      float bv = bias[col];
#pragma unroll
      for (int v = 0; v < 8; ++v) {
        int m = mt*16 + v + 8*g;
        out[(size_t)m*Cc + col] = accs[dt][v] + bv;
      }
    }
  }
}

// ---------------- host-side launcher ----------------
extern "C" void kernel_launch(void* const* d_in, const int* in_sizes, int n_in,
                              void* d_out, int out_size, void* d_ws, size_t ws_size,
                              hipStream_t stream) {
  (void)in_sizes; (void)n_in; (void)out_size; (void)ws_size;
  const float* x       = (const float*)d_in[0];
  const float* w_qkv   = (const float*)d_in[1];
  const float* w_proj  = (const float*)d_in[2];
  const float* b_proj  = (const float*)d_in[3];
  const float* ucb_cnt = (const float*)d_in[4];
  const int*   counter = (const int*)d_in[5];
  const int*   enabled = (const int*)d_in[6];
  float* out    = (float*)d_out;
  float* sdelta = out + (size_t)Bz*Nn*Cc;

  char* ws = (char*)d_ws;
  size_t off = 0;
  auto alloc = [&](size_t bytes) -> char* {
    char* p = ws + off;
    off = (off + bytes + 255) & ~(size_t)255;
    return p;
  };
  // zero-initialized region (padded q/k/vT rows, colsum accumulators)
  bf16*     qA      = (bf16*)    alloc((size_t)BH*NPQ*HDm*2);
  bf16*     kA      = (bf16*)    alloc((size_t)BH*NPK*HDm*2);
  bf16*     vT      = (bf16*)    alloc((size_t)BH*HDm*NPK*2);
  float*    colsum  = (float*)   alloc((size_t)Bz*Nn*4);
  size_t zeroBytes  = off;
  // write-before-read region
  bf16*     xb      = (bf16*)    alloc((size_t)MROWS*Cc*2);
  uint32_t* wqkvpk  = (uint32_t*)alloc((size_t)(Cc/2)*(3*Cc)*4);
  uint32_t* wprojpk = (uint32_t*)alloc((size_t)(Cc/2)*Cc*4);
  bf16*     P       = (bf16*)    alloc((size_t)BH*NPQ*NPK*2);
  float*    lbuf    = (float*)   alloc((size_t)BH*NPQ*4);
  float*    keep    = (float*)   alloc((size_t)Bz*NPQ*4);
  uint32_t* kmask   = (uint32_t*)alloc((size_t)Bz*NPK2*4);
  bf16*     ctx     = (bf16*)    alloc((size_t)MROWS*Cc*2);

  k_fill0   <<<2048, 256, 0, stream>>>((uint32_t*)ws, zeroBytes/4);
  k_cvt_bf16<<<2048, 256, 0, stream>>>(x, xb, (size_t)MROWS*Cc);
  k_cvt_pack<<<2048, 256, 0, stream>>>(w_qkv,  wqkvpk,  Cc/2, 3*Cc);
  k_cvt_pack<<<1024, 256, 0, stream>>>(w_proj, wprojpk, Cc/2, Cc);
  k_gemm_qkv<<<36*73, 256, 0, stream>>>(xb, wqkvpk, qA, kA, vT);
  k_attn_pass1<<<(BH*37 + 7)/8, 256, 0, stream>>>(qA, kA, P, lbuf);
  k_colsum  <<<BH*10, 256, 0, stream>>>(P, lbuf, colsum);
  k_topk    <<<Bz, 256, 0, stream>>>(colsum, ucb_cnt, counter, enabled, keep, kmask, sdelta);
  k_av      <<<(BH*37 + 7)/8, 256, 0, stream>>>(P, lbuf, kmask, keep, vT, ctx, counter, enabled);
  k_gemm_proj<<<12*73, 256, 0, stream>>>(ctx, wprojpk, b_proj, out);
}
